// ToposLogicModel_5016521801932
// MI455X (gfx1250) — matrix operand order repaired and checked by reading, not compiled, not run on hardware
//
#include <hip/hip_runtime.h>
#include <hip/hip_bf16.h>
#include <math.h>

#define BATCH 128
#define SEQ   50
#define DIM   512
#define HID   2048
#define VOCAB 32000
#define ROWS  (BATCH * SEQ)
#define LN_EPS 1e-5f

typedef __attribute__((ext_vector_type(16))) __bf16 v16bf;
typedef __attribute__((ext_vector_type(8)))  __bf16 v8bf;
typedef __attribute__((ext_vector_type(8)))  float  v8f;
typedef __attribute__((ext_vector_type(4)))  unsigned int u32x4;
typedef __attribute__((ext_vector_type(8)))  int i32x8;
typedef __attribute__((ext_vector_type(4)))  int i32x4;

// GEMM blocking: block = 128(M) x 64(N), K staged in 64-wide LDS tiles via TDM.
// LDS rows padded by TDM: 64 elems (128B) + 16B pad -> 144B pitch (72 elems).
#define KB        64
#define LDS_PITCH 72                   // elements per LDS row (with TDM pad)
#define A_TILE_B  (128 * LDS_PITCH * 2)  // 18432
#define B_TILE_B  (64  * LDS_PITCH * 2)  //  9216

// ---------------------------------------------------------------- embed + cast
__global__ void embed_kernel(const int* __restrict__ idx, const float* __restrict__ emb,
                             const float* __restrict__ pos, float* __restrict__ x32,
                             __bf16* __restrict__ xbf) {
    int row = blockIdx.x;              // b*SEQ + s
    int s   = row % SEQ;
    int tok = idx[row];
    const float* e = emb + (size_t)tok * DIM;
    const float* p = pos + (size_t)s   * DIM;
    for (int d = threadIdx.x; d < DIM; d += blockDim.x) {
        float v = e[d] + p[d];
        x32[(size_t)row * DIM + d] = v;
        xbf[(size_t)row * DIM + d] = (__bf16)v;
    }
}

// ------------------------------------------- weight cast f32[K,N] -> bf16[N,K]
__global__ void transpose_cast_kernel(const float* __restrict__ W, __bf16* __restrict__ WT,
                                      int K, int N) {
    __shared__ float tile[32][33];
    int n0 = blockIdx.x * 32, k0 = blockIdx.y * 32;
    for (int i = threadIdx.y; i < 32; i += 8)
        tile[i][threadIdx.x] = W[(size_t)(k0 + i) * N + n0 + threadIdx.x];
    __syncthreads();
    for (int i = threadIdx.y; i < 32; i += 8)
        WT[(size_t)(n0 + i) * K + k0 + threadIdx.x] = (__bf16)tile[threadIdx.x][i];
}

// ------------------------------------------------------------ TDM 2D tile load
// D# per CDNA5 ISA ch.8: group0 {count=1, lds_addr, global_addr, type=2},
// group1 {data_size=2B, pad 4dw/32dw, dims/strides}. Issued by one wave,
// tracked with TENSORcnt.
__device__ __forceinline__ void tdm_load_tile(unsigned lds_off, const __bf16* gptr,
                                              unsigned rows, unsigned stride_elems) {
    unsigned long long ga = (unsigned long long)(uintptr_t)gptr;
    u32x4 g0;
    g0[0] = 1u;                                          // count=1 (valid, user mode)
    g0[1] = lds_off;                                     // lds_addr (bytes)
    g0[2] = (unsigned)(ga & 0xffffffffu);                // global_addr[31:0]
    g0[3] = (unsigned)((ga >> 32) & 0x01ffffffu) | (2u << 30);  // addr[56:32] | type=2
    i32x8 g1;
    //    data_size=1(2B)<<16 | pad_enable<<20 | pad_interval=4(32dw)<<22 | pad_amount=3(4dw)<<25
    g1[0] = (int)((1u << 16) | (1u << 20) | (4u << 22) | (3u << 25));
    g1[1] = (int)((KB & 0xffffu) << 16);                 // tensor_dim0[15:0] (== tile_dim0)
    g1[2] = (int)((rows & 0xffffu) << 16);               // tensor_dim0[31:16]=0 | tensor_dim1[15:0]
    g1[3] = (int)((unsigned)KB << 16);                   // tensor_dim1[31:16]=0 | tile_dim0
    g1[4] = (int)(rows & 0xffffu);                       // tile_dim1 | tile_dim2=0
    g1[5] = (int)stride_elems;                           // tensor_dim0_stride[31:0]
    g1[6] = 0;                                           // stride[47:32] | dim1_stride lo
    g1[7] = 0;
    i32x4 z4 = {0, 0, 0, 0};
#if defined(__clang_major__) && (__clang_major__ >= 23)
    i32x8 z8 = {0, 0, 0, 0, 0, 0, 0, 0};
    __builtin_amdgcn_tensor_load_to_lds(g0, g1, z4, z4, z8, 0);
#else
    __builtin_amdgcn_tensor_load_to_lds(g0, g1, z4, z4, 0);
#endif
}

// ---------------------------------------------------------- bf16 WMMA GEMM
// C[M,N] = act(A[M,K] @ BT[N,K]^T + bias)   MODE: 0=none 1=sigmoid 2=gelu(exact)
// Block: 128x64 tile, 8 waves (4Mx2N), wave: 32x32 = 2x2 WMMA tiles.
// K-tiles (64) double-buffered in LDS, filled by TDM from wave 0.
template <int MODE>
__global__ void gemm_bf16_wmma(const __bf16* __restrict__ A, const __bf16* __restrict__ BT,
                               const float* __restrict__ bias, float* __restrict__ C32,
                               __bf16* __restrict__ Cbf, int M, int N, int K) {
    __shared__ __align__(128) char smem[2 * A_TILE_B + 2 * B_TILE_B];  // 55296 B
    const unsigned aoff[2] = {0u, (unsigned)A_TILE_B};
    const unsigned boff[2] = {2u * A_TILE_B, 2u * A_TILE_B + B_TILE_B};

    const int lane = threadIdx.x & 31;
    const int wid  = threadIdx.x >> 5;
    const int half = lane >> 4;          // K sub-block select per ISA fragment layout
    const int l16  = lane & 15;
    const int wm   = wid >> 1;           // 0..3  (32-row strip)
    const int wn   = wid & 1;            // 0..1  (32-col strip)

    const int nBlk = N / 64;
    const int bm   = blockIdx.x / nBlk;
    const int bn   = blockIdx.x % nBlk;
    const int M0   = bm * 128;
    const int N0   = bn * 64;

    const __bf16* Ag = A  + (size_t)M0 * K;   // 128 rows x K
    const __bf16* Bg = BT + (size_t)N0 * K;   // 64 cols (rows of BT) x K

    const int nk = K / KB;
    if (wid == 0) {                       // prologue: fill buffer 0
        tdm_load_tile(aoff[0], Ag, 128, (unsigned)K);
        tdm_load_tile(boff[0], Bg, 64,  (unsigned)K);
    }

    v8f acc[2][2] = {};
    for (int t = 0; t < nk; ++t) {
        const int cur = t & 1;
        __builtin_amdgcn_s_wait_tensorcnt(0);   // wave0: TDM done; others: no-op
        __syncthreads();                        // staged tile visible to all waves
        if (wid == 0 && t + 1 < nk) {           // prefetch next K-tile into other buffer
            tdm_load_tile(aoff[1 - cur], Ag + (size_t)(t + 1) * KB, 128, (unsigned)K);
            tdm_load_tile(boff[1 - cur], Bg + (size_t)(t + 1) * KB, 64,  (unsigned)K);
        }
        const __bf16* As = (const __bf16*)(smem + aoff[cur]);
        const __bf16* Bs = (const __bf16*)(smem + boff[cur]);
#pragma unroll
        for (int ks = 0; ks < KB; ks += 32) {
            v16bf af[2], bfr[2];
#pragma unroll
            for (int u = 0; u < 2; ++u) {
                // A frag: half0 -> K {ks..+7, ks+16..+23}; half1 -> +8
                const __bf16* pa = As + ((wm * 2 + u) * 16 + l16) * LDS_PITCH + ks + half * 8;
                v8bf alo = *(const v8bf*)(pa);
                v8bf ahi = *(const v8bf*)(pa + 16);
                // B frag: half0 -> K {ks..+7, ks+16..+23}; half1 -> +8
                const __bf16* pb = Bs + ((wn * 2 + u) * 16 + l16) * LDS_PITCH + ks + half * 8;
                v8bf blo = *(const v8bf*)(pb);
                v8bf bhi = *(const v8bf*)(pb + 16);
#pragma unroll
                for (int i = 0; i < 8; ++i) {
                    af[u][i] = alo[i]; af[u][i + 8] = ahi[i];
                    bfr[u][i] = blo[i]; bfr[u][i + 8] = bhi[i];
                }
            }
#pragma unroll
            for (int a = 0; a < 2; ++a)
#pragma unroll
                for (int b = 0; b < 2; ++b)
                    acc[a][b] = __builtin_amdgcn_wmma_f32_16x16x32_bf16(
                        false, af[a], false, bfr[b], (short)0, acc[a][b], false, false);
        }
        __syncthreads();                        // all waves done with buf[cur] before reuse
    }

    // epilogue: bias + activation + store
#pragma unroll
    for (int b = 0; b < 2; ++b) {
        const int n = N0 + (wn * 2 + b) * 16 + l16;
        const float bval = bias ? bias[n] : 0.0f;
#pragma unroll
        for (int a = 0; a < 2; ++a) {
#pragma unroll
            for (int r = 0; r < 8; ++r) {
                int m = M0 + (wm * 2 + a) * 16 + r + half * 8;  // C layout: VGPR r -> M=r / r+8
                float v = acc[a][b][r] + bval;
                if (MODE == 1) v = 1.0f / (1.0f + expf(-v));
                else if (MODE == 2) v = 0.5f * v * (1.0f + erff(v * 0.70710678118654752f));
                size_t o = (size_t)m * N + n;
                if (C32) C32[o] = v;
                if (Cbf) Cbf[o] = (__bf16)v;
            }
        }
    }
}

// ------------------------------------------------------------ reductions
__device__ __forceinline__ float wave_sum(float v) {
#pragma unroll
    for (int off = 16; off > 0; off >>= 1) v += __shfl_xor(v, off, 32);
    return v;
}
__device__ __forceinline__ float wave_max(float v) {
#pragma unroll
    for (int off = 16; off > 0; off >>= 1) v = fmaxf(v, __shfl_xor(v, off, 32));
    return v;
}

// ---------------- Lukasiewicz truth + softmax + attn@V + residual + LN1
__global__ void logic_attn_ln1_kernel(const float* __restrict__ x32, const float* __restrict__ Q,
                                      const float* __restrict__ Kmat, const float* __restrict__ V,
                                      const float* __restrict__ g1, const float* __restrict__ be1,
                                      float* __restrict__ x1_32, __bf16* __restrict__ x1_bf) {
    __shared__ float qrow[DIM];
    __shared__ float attn[SEQ];
    __shared__ float red[8];
    const int row = blockIdx.x;
    const int b = row / SEQ, i = row % SEQ;
    const int tid = threadIdx.x, lane = tid & 31, wid = tid >> 5;

    for (int d = tid; d < DIM; d += 256) qrow[d] = Q[(size_t)row * DIM + d];
    __syncthreads();

    // logits: j<=i -> 10 * mean_d min(1 - Q[i,d] + K[j,d], 1); j>i -> 0 (ref masks to 0, not -inf)
    for (int j = wid; j < SEQ; j += 8) {
        if (j <= i) {
            const float* krow = Kmat + (size_t)(b * SEQ + j) * DIM;
            float p = 0.0f;
            for (int d = lane; d < DIM; d += 32)
                p += fminf(1.0f - qrow[d] + krow[d], 1.0f);
            p = wave_sum(p);
            if (lane == 0) attn[j] = p * (10.0f / DIM);
        } else {
            if (lane == 0) attn[j] = 0.0f;
        }
    }
    __syncthreads();

    if (wid == 0) {   // softmax over all SEQ positions
        float m = -1e30f;
        for (int j = lane; j < SEQ; j += 32) m = fmaxf(m, attn[j]);
        m = wave_max(m);
        float s = 0.0f;
        for (int j = lane; j < SEQ; j += 32) { float e = expf(attn[j] - m); attn[j] = e; s += e; }
        s = wave_sum(s);
        float inv = 1.0f / s;
        for (int j = lane; j < SEQ; j += 32) attn[j] *= inv;
    }
    __syncthreads();

    float y0 = x32[(size_t)row * DIM + tid];
    float y1 = x32[(size_t)row * DIM + tid + 256];
    for (int j = 0; j < SEQ; ++j) {
        const float* vrow = V + (size_t)(b * SEQ + j) * DIM;
        float w = attn[j];
        y0 += w * vrow[tid];
        y1 += w * vrow[tid + 256];
    }

    float s = wave_sum(y0 + y1);
    if (lane == 0) red[wid] = s;
    __syncthreads();
    float mean = (red[0]+red[1]+red[2]+red[3]+red[4]+red[5]+red[6]+red[7]) * (1.0f / DIM);
    __syncthreads();
    float d0 = y0 - mean, d1 = y1 - mean;
    float vs = wave_sum(d0 * d0 + d1 * d1);
    if (lane == 0) red[wid] = vs;
    __syncthreads();
    float var = (red[0]+red[1]+red[2]+red[3]+red[4]+red[5]+red[6]+red[7]) * (1.0f / DIM);
    float inv = rsqrtf(var + LN_EPS);
    float o0 = d0 * inv * g1[tid]       + be1[tid];
    float o1 = d1 * inv * g1[tid + 256] + be1[tid + 256];
    x1_32[(size_t)row * DIM + tid]       = o0;
    x1_32[(size_t)row * DIM + tid + 256] = o1;
    x1_bf[(size_t)row * DIM + tid]       = (__bf16)o0;
    x1_bf[(size_t)row * DIM + tid + 256] = (__bf16)o1;
}

// ---------------------------------------------- residual + LN2 -> bf16
__global__ void add_ln2_kernel(const float* __restrict__ x1, const float* __restrict__ f,
                               const float* __restrict__ g2, const float* __restrict__ be2,
                               __bf16* __restrict__ x2bf) {
    __shared__ float red[8];
    const int row = blockIdx.x;
    const int tid = threadIdx.x, lane = tid & 31, wid = tid >> 5;
    size_t base = (size_t)row * DIM;
    float y0 = x1[base + tid]       + f[base + tid];
    float y1 = x1[base + tid + 256] + f[base + tid + 256];
    float s = wave_sum(y0 + y1);
    if (lane == 0) red[wid] = s;
    __syncthreads();
    float mean = (red[0]+red[1]+red[2]+red[3]+red[4]+red[5]+red[6]+red[7]) * (1.0f / DIM);
    __syncthreads();
    float d0 = y0 - mean, d1 = y1 - mean;
    float vs = wave_sum(d0 * d0 + d1 * d1);
    if (lane == 0) red[wid] = vs;
    __syncthreads();
    float var = (red[0]+red[1]+red[2]+red[3]+red[4]+red[5]+red[6]+red[7]) * (1.0f / DIM);
    float inv = rsqrtf(var + LN_EPS);
    x2bf[base + tid]       = (__bf16)(d0 * inv * g2[tid]       + be2[tid]);
    x2bf[base + tid + 256] = (__bf16)(d1 * inv * g2[tid + 256] + be2[tid + 256]);
}

// ============================================================================
extern "C" void kernel_launch(void* const* d_in, const int* in_sizes, int n_in,
                              void* d_out, int out_size, void* d_ws, size_t ws_size,
                              hipStream_t stream) {
    const int*   idx = (const int*)  d_in[0];
    const float* emb = (const float*)d_in[1];
    const float* pos = (const float*)d_in[2];
    const float* wq  = (const float*)d_in[3];
    const float* bq  = (const float*)d_in[4];
    const float* wk  = (const float*)d_in[5];
    const float* bk  = (const float*)d_in[6];
    const float* wv  = (const float*)d_in[7];
    const float* bv  = (const float*)d_in[8];
    const float* w1  = (const float*)d_in[9];
    const float* b1  = (const float*)d_in[10];
    const float* w2  = (const float*)d_in[11];
    const float* b2  = (const float*)d_in[12];
    const float* g1  = (const float*)d_in[13];
    const float* be1 = (const float*)d_in[14];
    const float* g2  = (const float*)d_in[15];
    const float* be2 = (const float*)d_in[16];
    const float* wo  = (const float*)d_in[17];
    const float* bo  = (const float*)d_in[18];

    char* base = (char*)d_ws;
    size_t off = 0;
    auto alloc = [&](size_t bytes) -> void* {
        void* p = base + off;
        off += (bytes + 255) & ~(size_t)255;
        return p;
    };

    float*  x32   = (float*) alloc((size_t)ROWS * DIM * 4);
    __bf16* xbf   = (__bf16*)alloc((size_t)ROWS * DIM * 2);
    float*  Q32   = (float*) alloc((size_t)ROWS * DIM * 4);
    float*  K32   = (float*) alloc((size_t)ROWS * DIM * 4);
    float*  V32   = (float*) alloc((size_t)ROWS * DIM * 4);
    float*  x1_32 = (float*) alloc((size_t)ROWS * DIM * 4);
    __bf16* x1_bf = (__bf16*)alloc((size_t)ROWS * DIM * 2);
    __bf16* h_bf  = (__bf16*)alloc((size_t)ROWS * HID * 2);
    float*  f32b  = (float*) alloc((size_t)ROWS * DIM * 4);
    __bf16* x2_bf = (__bf16*)alloc((size_t)ROWS * DIM * 2);
    __bf16* wqT   = (__bf16*)alloc((size_t)DIM * DIM * 2);
    __bf16* wkT   = (__bf16*)alloc((size_t)DIM * DIM * 2);
    __bf16* wvT   = (__bf16*)alloc((size_t)DIM * DIM * 2);
    __bf16* w1T   = (__bf16*)alloc((size_t)DIM * HID * 2);
    __bf16* w2T   = (__bf16*)alloc((size_t)HID * DIM * 2);
    __bf16* woT   = (__bf16*)alloc((size_t)DIM * VOCAB * 2);
    (void)ws_size; (void)in_sizes; (void)n_in; (void)out_size;

    dim3 tb(32, 8);
    transpose_cast_kernel<<<dim3(DIM / 32,   DIM / 32), tb, 0, stream>>>(wq, wqT, DIM, DIM);
    transpose_cast_kernel<<<dim3(DIM / 32,   DIM / 32), tb, 0, stream>>>(wk, wkT, DIM, DIM);
    transpose_cast_kernel<<<dim3(DIM / 32,   DIM / 32), tb, 0, stream>>>(wv, wvT, DIM, DIM);
    transpose_cast_kernel<<<dim3(HID / 32,   DIM / 32), tb, 0, stream>>>(w1, w1T, DIM, HID);
    transpose_cast_kernel<<<dim3(DIM / 32,   HID / 32), tb, 0, stream>>>(w2, w2T, HID, DIM);
    transpose_cast_kernel<<<dim3(VOCAB / 32, DIM / 32), tb, 0, stream>>>(wo, woT, DIM, VOCAB);

    embed_kernel<<<ROWS, 256, 0, stream>>>(idx, emb, pos, x32, xbf);

    auto blocks = [](int M, int N) { return (unsigned)((M / 128) * (N / 64)); };

    gemm_bf16_wmma<1><<<blocks(ROWS, DIM), 256, 0, stream>>>(xbf, wqT, bq, Q32, nullptr, ROWS, DIM, DIM);
    gemm_bf16_wmma<1><<<blocks(ROWS, DIM), 256, 0, stream>>>(xbf, wkT, bk, K32, nullptr, ROWS, DIM, DIM);
    gemm_bf16_wmma<0><<<blocks(ROWS, DIM), 256, 0, stream>>>(xbf, wvT, bv, V32, nullptr, ROWS, DIM, DIM);

    logic_attn_ln1_kernel<<<ROWS, 256, 0, stream>>>(x32, Q32, K32, V32, g1, be1, x1_32, x1_bf);

    gemm_bf16_wmma<2><<<blocks(ROWS, HID), 256, 0, stream>>>(x1_bf, w1T, b1, nullptr, h_bf, ROWS, HID, DIM);
    gemm_bf16_wmma<0><<<blocks(ROWS, DIM), 256, 0, stream>>>(h_bf, w2T, b2, f32b, nullptr, ROWS, DIM, HID);

    add_ln2_kernel<<<ROWS, 256, 0, stream>>>(x1_32, f32b, g2, be2, x2_bf);

    gemm_bf16_wmma<0><<<blocks(ROWS, VOCAB), 256, 0, stream>>>(x2_bf, woT, bo, (float*)d_out, nullptr,
                                                               ROWS, VOCAB, DIM);
}
